// RRNetModel_87402584473684
// MI455X (gfx1250) — compile-verified, matrix operand order checked by reading, not run on hardware
//
#include <hip/hip_runtime.h>
#include <math.h>

// ---------------- types / constants ----------------
typedef __bf16 bf16_t;
typedef __attribute__((ext_vector_type(16))) __bf16 v16bf;
typedef __attribute__((ext_vector_type(8)))  float  v8f;

#define TT    128
#define BB    32
#define NTOK  32000
#define NINP  512
#define NHID  1024
#define SMAXD 2
#define GDIM  4096            // 4*NHID
#define KALL  2560            // NINP + NHID(si) + NHID(h)
#define NALL  12288           // 3*GDIM
#define DECN  ((size_t)TT*BB*NTOK)
#define DEC_LDS_STRIDE 520    // 512 + 8 bf16 pad (keeps 8B alignment, spreads banks)

// ---------------- helpers ----------------
__device__ __forceinline__ bf16_t f2bf(float f) {
  unsigned int u = __float_as_uint(f);
  unsigned int r = (u + 0x7FFFu + ((u >> 16) & 1u)) >> 16;
  unsigned short s = (unsigned short)r;
  bf16_t o; __builtin_memcpy(&o, &s, 2); return o;
}
__device__ __forceinline__ float bf2f(bf16_t b) {
  unsigned short s; __builtin_memcpy(&s, &b, 2);
  return __uint_as_float(((unsigned int)s) << 16);
}
__device__ __forceinline__ float sigf(float x) { return 1.0f / (1.0f + __expf(-x)); }

__device__ __forceinline__ unsigned int mixh(unsigned int a) {
  a ^= a >> 16; a *= 0x7feb352dU; a ^= a >> 15; a *= 0x846ca68bU; a ^= a >> 16; return a;
}

// Load a 16x32 bf16 tile (rows x K) from row-major memory into the CDNA5
// WMMA A/B per-lane layout: lane L holds row L%16; VGPR i holds K pair
// k = (i<4?0:16) + (i&3)*2 + (L/16)*8.
__device__ __forceinline__ v16bf load_tile(const bf16_t* p, int ld, int lane) {
  const int row  = lane & 15;
  const int half = (lane >> 4) & 1;
  const bf16_t* rp = p + (size_t)row * ld + half * 8;
  union { v16bf v; unsigned int u[8]; } t;
#pragma unroll
  for (int i = 0; i < 8; ++i) {
    const int k = ((i < 4) ? 0 : 16) + (i & 3) * 2;
    t.u[i] = *(const unsigned int*)(rp + k);
  }
  return t.v;
}

// ---------------- prep kernels ----------------
// Pack [rWih|0|rWhh ; sWih|0|sWhh ; mWih|mWhh] into Wall (NALL x KALL) bf16.
__global__ void prep_wall_kernel(const float* __restrict__ rWih, const float* __restrict__ rWhh, const float* __restrict__ rb,
                                 const float* __restrict__ sWih, const float* __restrict__ sWhh, const float* __restrict__ sb,
                                 const float* __restrict__ mWih, const float* __restrict__ mWhh, const float* __restrict__ mb,
                                 bf16_t* __restrict__ Wall, float* __restrict__ biasAll) {
  size_t idx = (size_t)blockIdx.x * 256 + threadIdx.x;
  if (idx >= (size_t)NALL * KALL) return;
  int n = (int)(idx / KALL);
  int c = (int)(idx % KALL);
  int cell = n >> 12;
  int j = n & (GDIM - 1);
  float v = 0.0f;
  if (cell == 0) {
    if (c < NINP) v = rWih[(size_t)j * NINP + c];
    else if (c >= NINP + NHID) v = rWhh[(size_t)j * NHID + (c - NINP - NHID)];
  } else if (cell == 1) {
    if (c < NINP) v = sWih[(size_t)j * NINP + c];
    else if (c >= NINP + NHID) v = sWhh[(size_t)j * NHID + (c - NINP - NHID)];
  } else {
    if (c < NINP + NHID) v = mWih[(size_t)j * (NINP + NHID) + c];
    else v = mWhh[(size_t)j * NHID + (c - NINP - NHID)];
  }
  Wall[idx] = f2bf(v);
  if (c == 0) biasAll[n] = (cell == 0) ? rb[j] : (cell == 1) ? sb[j] : mb[j];
}

__global__ void convert_bf16_kernel(const float* __restrict__ src, bf16_t* __restrict__ dst, int n) {
  int idx = blockIdx.x * 256 + threadIdx.x;
  if (idx < n) dst[idx] = f2bf(src[idx]);
}

__global__ void embed_kernel(const int* __restrict__ tokens, const float* __restrict__ embW,
                             bf16_t* __restrict__ Xbf) {
  int idx = blockIdx.x * 256 + threadIdx.x;      // TT*BB*NINP
  if (idx >= TT * BB * NINP) return;
  int row = idx / NINP, c = idx % NINP;
  int tok = tokens[row];
  Xbf[idx] = f2bf(embW[(size_t)tok * NINP + c]);
}

__global__ void init_state_kernel(const float* __restrict__ h0, const float* __restrict__ c0,
                                  float* __restrict__ h, float* __restrict__ c,
                                  float* __restrict__ stack, int* __restrict__ depth) {
  int idx = blockIdx.x * 256 + threadIdx.x;
  if (idx < BB * NHID) { h[idx] = h0[idx]; c[idx] = c0[idx]; }
  if (idx < BB * SMAXD * NHID) stack[idx] = 0.0f;
  if (idx < BB) depth[idx] = 0;
}

// ---------------- per-step kernels ----------------
// Fused: block b (of 32) both assembles A row b = [hi(512)|si(1024)|h(1024)]
// in bf16 and runs the policy MLP + layernorm + gumbel sampling for row b.
// Both parts only read (h, stack, depth, Xbf) and write disjoint outputs.
__global__ void policy_assemble_kernel(int t, const float* __restrict__ h, const bf16_t* __restrict__ Xbf,
                                       const float* __restrict__ stack, const int* __restrict__ depth,
                                       const float* __restrict__ piW1, const float* __restrict__ pib1,
                                       const float* __restrict__ lng, const float* __restrict__ lnb,
                                       const float* __restrict__ piW2, const float* __restrict__ pib2,
                                       int* __restrict__ actions, bf16_t* __restrict__ Abuf) {
  __shared__ float zs[NINP];
  __shared__ float red[256];
  __shared__ float mu_s, var_s, lg[3];
  const int b = blockIdx.x, tid = threadIdx.x;
  const float* hb = h + (size_t)b * NHID;
  const bf16_t* xi = Xbf + (size_t)(t * BB + b) * NINP;
  const int d_b = depth[b];

  // ---- assemble A row b ----
  {
    int slot = d_b - 1; slot = slot < 0 ? 0 : (slot > SMAXD - 1 ? SMAXD - 1 : slot);
    const float* topp = stack + ((size_t)b * SMAXD + slot) * NHID;
    bf16_t* arow = Abuf + (size_t)b * KALL;
    for (int c = tid; c < KALL; c += 256) {
      float v;
      if (c < NINP) v = bf2f(xi[c]);
      else if (c < NINP + NHID) v = (d_b > 0) ? topp[c - NINP] : 0.0f;
      else v = hb[c - NINP - NHID];
      arow[c] = f2bf(v);
    }
  }

  // ---- policy MLP ----
  for (int j = tid; j < NINP; j += 256) {
    const float* w = piW1 + (size_t)j * (NHID + NINP);
    float acc = pib1[j];
    for (int cidx = 0; cidx < NHID; ++cidx) acc += hb[cidx] * w[cidx];
    for (int cidx = 0; cidx < NINP; ++cidx) acc += bf2f(xi[cidx]) * w[NHID + cidx];
    zs[j] = acc;
  }
  __syncthreads();
  float s = 0.f, s2 = 0.f;
  for (int j = tid; j < NINP; j += 256) { float z = zs[j]; s += z; s2 += z * z; }
  red[tid] = s; __syncthreads();
  for (int off = 128; off > 0; off >>= 1) { if (tid < off) red[tid] += red[tid + off]; __syncthreads(); }
  if (tid == 0) mu_s = red[0] / NINP;
  __syncthreads();
  red[tid] = s2; __syncthreads();
  for (int off = 128; off > 0; off >>= 1) { if (tid < off) red[tid] += red[tid + off]; __syncthreads(); }
  if (tid == 0) var_s = red[0] / NINP - mu_s * mu_s;
  __syncthreads();
  const float mu = mu_s;
  const float inv = rsqrtf(var_s + 1e-5f);
  for (int j = tid; j < NINP; j += 256) {
    float z = (zs[j] - mu) * inv * lng[j] + lnb[j];
    zs[j] = fmaxf(z, 0.0f);
  }
  __syncthreads();
  for (int o = 0; o < 3; ++o) {
    float l = 0.f;
    for (int j = tid; j < NINP; j += 256) l += zs[j] * piW2[(size_t)o * NINP + j];
    red[tid] = l; __syncthreads();
    for (int off = 128; off > 0; off >>= 1) { if (tid < off) red[tid] += red[tid + off]; __syncthreads(); }
    if (tid == 0) lg[o] = red[0] + pib2[o];
    __syncthreads();
  }
  if (tid == 0) {
    float ls[3];
    ls[0] = (d_b >= SMAXD) ? -1e30f : lg[0];
    ls[1] = lg[1];
    ls[2] = (d_b == 0) ? -1e30f : lg[2];
    int best = 1; float bv = -1e38f;
#pragma unroll
    for (int k = 0; k < 3; ++k) {
      unsigned int hbits = mixh(0x42u ^ mixh((unsigned)t * 0x9E3779B9u) ^ mixh((unsigned)b * 0x85EBCA6Bu + k));
      float u = (float)(hbits >> 8) * (1.0f / 16777216.0f) + 1e-12f;
      float g = ls[k] - __logf(-__logf(u));
      if (g > bv) { bv = g; best = k; }
    }
    actions[b] = best;
  }
}

// Gates(32 x NALL) = A(32 x KALL) @ Wall^T + biasAll   -- WMMA bf16, f32 acc
// Latency-bound (M=32, sequential over T): keep max N-parallelism, 1 tile/wave.
__global__ void __launch_bounds__(256) step_gemm_kernel(const bf16_t* __restrict__ Abuf,
                                                        const bf16_t* __restrict__ Wall,
                                                        const float* __restrict__ biasAll,
                                                        float* __restrict__ Gates) {
  const int wave = threadIdx.x >> 5, lane = threadIdx.x & 31;
  const int ntile = blockIdx.x * 8 + wave;         // 0..767
  const bf16_t* Bp = Wall + (size_t)ntile * 16 * KALL;
  v8f acc0 = {}; v8f acc1 = {};
  for (int k = 0; k < KALL; k += 32) {
    v16bf a0 = load_tile(Abuf + k, KALL, lane);
    v16bf a1 = load_tile(Abuf + (size_t)16 * KALL + k, KALL, lane);
    v16bf bt = load_tile(Bp + k, KALL, lane);
    acc0 = __builtin_amdgcn_wmma_f32_16x16x32_bf16(false, a0, false, bt, (short)0, acc0, false, false);
    acc1 = __builtin_amdgcn_wmma_f32_16x16x32_bf16(false, a1, false, bt, (short)0, acc1, false, false);
  }
  const int n = ntile * 16 + (lane & 15);
  const int half = lane >> 4;
  const float bias = biasAll[n];
#pragma unroll
  for (int v = 0; v < 8; ++v) {
    int m = v + 8 * half;
    Gates[(size_t)m * NALL + n] = acc0[v] + bias;
    Gates[(size_t)(m + 16) * NALL + n] = acc1[v] + bias;
  }
}

// Fused LSTM pointwise + stack push + depth update. One block per batch row:
// all stack writes for row b (which must observe the OLD depth) complete
// inside the block before thread 0 bumps depth[b] after the barrier.
__global__ void pointwise_kernel(int t, const float* __restrict__ Gates,
                                 float* __restrict__ h, float* __restrict__ c,
                                 float* __restrict__ stack, int* __restrict__ depth,
                                 const int* __restrict__ actions, bf16_t* __restrict__ outs) {
  const int b = blockIdx.x;                        // 32 blocks
  const int tid = threadIdx.x;
  const int a = actions[b];
  const int d = depth[b];
  for (int j = tid; j < NHID; j += 256) {
    const size_t idx = (size_t)b * NHID + j;
    const float cp = c[idx];
    float hc[3], cc[3];
#pragma unroll
    for (int cell = 0; cell < 3; ++cell) {
      const float* g = Gates + (size_t)b * NALL + (size_t)cell * GDIM;
      float gi = sigf(g[j]);
      float gf = sigf(g[NHID + j]);
      float gg = tanhf(g[2 * NHID + j]);
      float go = sigf(g[3 * NHID + j]);
      float c2 = gf * cp + gi * gg;
      cc[cell] = c2;
      hc[cell] = go * tanhf(c2);
    }
    // cell order in Wall: 0=r(reduce), 1=s(shift), 2=m(merge); action 0->s, 1->r, 2->m
    float hn = (a == 0) ? hc[1] : (a == 1) ? hc[0] : hc[2];
    float cn = (a == 0) ? cc[1] : (a == 1) ? cc[0] : cc[2];
    if (a == 0 && d < SMAXD) stack[((size_t)b * SMAXD + d) * NHID + j] = hc[1];
    h[idx] = hn;
    c[idx] = cn;
    outs[(size_t)(t * BB + b) * NHID + j] = f2bf(hn);
  }
  __syncthreads();
  if (tid == 0) depth[b] = d + (a == 0 ? 1 : 0) - (a == 2 ? 1 : 0);
}

// ---------------- epilogue GEMMs ----------------
// y = tanh((outs @ predW^T + predb) / sqrt(1+eps) * bn_g + bn_b) -> bf16
__global__ void __launch_bounds__(256) pred_gemm_kernel(const bf16_t* __restrict__ outs,
                                                        const bf16_t* __restrict__ predW,
                                                        const float* __restrict__ predb,
                                                        const float* __restrict__ bng,
                                                        const float* __restrict__ bnb,
                                                        bf16_t* __restrict__ Ybf) {
  const int wave = threadIdx.x >> 5, lane = threadIdx.x & 31;
  const int ntile = blockIdx.x * 8 + wave;         // 0..31
  const int mtile = blockIdx.y;                    // 0..255
  const bf16_t* Ap = outs + (size_t)mtile * 16 * NHID;
  const bf16_t* Bp = predW + (size_t)ntile * 16 * NHID;
  v8f acc = {};
  for (int k = 0; k < NHID; k += 32) {
    v16bf a  = load_tile(Ap + k, NHID, lane);
    v16bf bt = load_tile(Bp + k, NHID, lane);
    acc = __builtin_amdgcn_wmma_f32_16x16x32_bf16(false, a, false, bt, (short)0, acc, false, false);
  }
  const int n = ntile * 16 + (lane & 15);
  const int half = lane >> 4;
  const float inv = rsqrtf(1.0f + 1e-5f);
  const float g = bng[n], bb = bnb[n], pb = predb[n];
#pragma unroll
  for (int v = 0; v < 8; ++v) {
    int m = mtile * 16 + v + 8 * half;
    float val = tanhf((acc[v] + pb) * inv * g + bb);
    Ybf[(size_t)m * NINP + n] = f2bf(val);
  }
}

// dec = y @ decW^T + dec_b  -> f32 straight to d_out.
// Dominant kernel: stage the block's 32x512 A panel in LDS once (33KB of the
// 320KB WGP LDS), then 2x2 register blocking: 4 WMMAs per (2 ds + 2 vmem)
// tile loads. A global traffic cut 8x; ds pipe co-issues with VMEM.
__global__ void __launch_bounds__(256) dec_gemm_kernel(const bf16_t* __restrict__ Ybf,
                                                       const bf16_t* __restrict__ decW,
                                                       const float* __restrict__ decb,
                                                       float* __restrict__ out) {
  __shared__ bf16_t As[32 * DEC_LDS_STRIDE];
  const int tid = threadIdx.x;
  const int wave = tid >> 5, lane = tid & 31;
  const int m0 = blockIdx.y * 32;                  // 32-row super-tile (128 total)

  // cooperative fill: 32 rows x 512 cols bf16, 8-byte units (2048 units / 256 thr)
  for (int u = tid; u < 32 * 64; u += 256) {
    int r = u >> 6, c8 = (u & 63) * 8;
    *(unsigned long long*)(As + (size_t)r * DEC_LDS_STRIDE + c8) =
        *(const unsigned long long*)(Ybf + (size_t)(m0 + r) * NINP + c8);
  }
  __syncthreads();

  const int nt0 = (blockIdx.x * 8 + wave) * 2;     // two adjacent n-tiles
  const bf16_t* Bp0 = decW + (size_t)nt0 * 16 * NINP;
  const bf16_t* Bp1 = decW + (size_t)(nt0 + 1) * 16 * NINP;
  v8f acc00 = {}, acc01 = {}, acc10 = {}, acc11 = {};
  for (int k = 0; k < NINP; k += 32) {
    __builtin_prefetch(Bp0 + k + 128, 0, 1);       // global_prefetch_b8 on weights
    __builtin_prefetch(Bp1 + k + 128, 0, 1);
    v16bf a0 = load_tile(As + k, DEC_LDS_STRIDE, lane);                               // rows m0..+15 (LDS)
    v16bf a1 = load_tile(As + (size_t)16 * DEC_LDS_STRIDE + k, DEC_LDS_STRIDE, lane); // rows +16..+31
    v16bf b0 = load_tile(Bp0 + k, NINP, lane);
    v16bf b1 = load_tile(Bp1 + k, NINP, lane);
    acc00 = __builtin_amdgcn_wmma_f32_16x16x32_bf16(false, a0, false, b0, (short)0, acc00, false, false);
    acc01 = __builtin_amdgcn_wmma_f32_16x16x32_bf16(false, a0, false, b1, (short)0, acc01, false, false);
    acc10 = __builtin_amdgcn_wmma_f32_16x16x32_bf16(false, a1, false, b0, (short)0, acc10, false, false);
    acc11 = __builtin_amdgcn_wmma_f32_16x16x32_bf16(false, a1, false, b1, (short)0, acc11, false, false);
  }
  const int n0 = nt0 * 16 + (lane & 15);
  const int n1 = n0 + 16;
  const int half = lane >> 4;
  const float bias0 = decb[n0], bias1 = decb[n1];
#pragma unroll
  for (int v = 0; v < 8; ++v) {
    int m = m0 + v + 8 * half;
    out[(size_t)m * NTOK + n0]        = acc00[v] + bias0;
    out[(size_t)m * NTOK + n1]        = acc01[v] + bias1;
    out[(size_t)(m + 16) * NTOK + n0] = acc10[v] + bias0;
    out[(size_t)(m + 16) * NTOK + n1] = acc11[v] + bias1;
  }
}

__global__ void copy_hc_kernel(const float* __restrict__ h, const float* __restrict__ c,
                               float* __restrict__ out) {
  int idx = blockIdx.x * 256 + threadIdx.x;
  if (idx < BB * NHID) out[DECN + idx] = h[idx];
  else if (idx < 2 * BB * NHID) out[DECN + idx] = c[idx - BB * NHID];
}

// ---------------- host launcher ----------------
extern "C" void kernel_launch(void* const* d_in, const int* in_sizes, int n_in,
                              void* d_out, int out_size, void* d_ws, size_t ws_size,
                              hipStream_t stream) {
  (void)in_sizes; (void)n_in; (void)out_size; (void)ws_size;
  const int*   tokens = (const int*)  d_in[0];
  const float* h0     = (const float*)d_in[1];
  const float* c0     = (const float*)d_in[2];
  const float* embW   = (const float*)d_in[3];
  const float* rWih   = (const float*)d_in[4];
  const float* rWhh   = (const float*)d_in[5];
  const float* rb     = (const float*)d_in[6];
  const float* sWih   = (const float*)d_in[7];
  const float* sWhh   = (const float*)d_in[8];
  const float* sb     = (const float*)d_in[9];
  const float* mWih   = (const float*)d_in[10];
  const float* mWhh   = (const float*)d_in[11];
  const float* mb     = (const float*)d_in[12];
  const float* piW1   = (const float*)d_in[13];
  const float* pib1   = (const float*)d_in[14];
  const float* lng    = (const float*)d_in[15];
  const float* lnb    = (const float*)d_in[16];
  const float* piW2   = (const float*)d_in[17];
  const float* pib2   = (const float*)d_in[18];
  const float* predW  = (const float*)d_in[19];
  const float* predb  = (const float*)d_in[20];
  const float* bng    = (const float*)d_in[21];
  const float* bnb    = (const float*)d_in[22];
  const float* decW   = (const float*)d_in[23];
  const float* decb   = (const float*)d_in[24];
  float* out = (float*)d_out;

  char* ws = (char*)d_ws;
  size_t off = 0;
  auto alloc = [&](size_t bytes) -> void* {
    void* p = ws + off;
    off += (bytes + 255) & ~(size_t)255;
    return p;
  };
  bf16_t* Wall    = (bf16_t*)alloc((size_t)NALL * KALL * 2);       // 62.9 MB
  float*  biasAll = (float*) alloc((size_t)NALL * 4);
  bf16_t* decWbf  = (bf16_t*)alloc((size_t)NTOK * NINP * 2);       // 32.8 MB
  bf16_t* predWbf = (bf16_t*)alloc((size_t)NINP * NHID * 2);       // 1.0 MB
  bf16_t* Xbf     = (bf16_t*)alloc((size_t)TT * BB * NINP * 2);    // 4.2 MB
  bf16_t* Abuf    = (bf16_t*)alloc((size_t)BB * KALL * 2);
  float*  Gates   = (float*) alloc((size_t)BB * NALL * 4);         // 1.6 MB
  float*  hState  = (float*) alloc((size_t)BB * NHID * 4);
  float*  cState  = (float*) alloc((size_t)BB * NHID * 4);
  float*  stackB  = (float*) alloc((size_t)BB * SMAXD * NHID * 4);
  int*    depth   = (int*)   alloc(BB * 4);
  int*    actions = (int*)   alloc(BB * 4);
  bf16_t* outsBf  = (bf16_t*)alloc((size_t)TT * BB * NHID * 2);    // 8.4 MB
  bf16_t* Ybf     = (bf16_t*)alloc((size_t)TT * BB * NINP * 2);    // 4.2 MB

  // --- prep ---
  prep_wall_kernel<<<(unsigned)(((size_t)NALL * KALL + 255) / 256), 256, 0, stream>>>(
      rWih, rWhh, rb, sWih, sWhh, sb, mWih, mWhh, mb, Wall, biasAll);
  convert_bf16_kernel<<<(NTOK * NINP + 255) / 256, 256, 0, stream>>>(decW, decWbf, NTOK * NINP);
  convert_bf16_kernel<<<(NINP * NHID + 255) / 256, 256, 0, stream>>>(predW, predWbf, NINP * NHID);
  embed_kernel<<<(TT * BB * NINP + 255) / 256, 256, 0, stream>>>(tokens, embW, Xbf);
  init_state_kernel<<<(BB * SMAXD * NHID + 255) / 256, 256, 0, stream>>>(h0, c0, hState, cState, stackB, depth);

  // --- sequential scan: 3 graph nodes per step ---
  for (int t = 0; t < TT; ++t) {
    policy_assemble_kernel<<<BB, 256, 0, stream>>>(t, hState, Xbf, stackB, depth,
                                                   piW1, pib1, lng, lnb, piW2, pib2, actions, Abuf);
    step_gemm_kernel<<<(NALL / 16) / 8, 256, 0, stream>>>(Abuf, Wall, biasAll, Gates);
    pointwise_kernel<<<BB, 256, 0, stream>>>(t, Gates, hState, cState, stackB, depth, actions, outsBf);
  }

  // --- epilogue ---
  pred_gemm_kernel<<<dim3((NINP / 16) / 8, (TT * BB) / 16), 256, 0, stream>>>(
      outsBf, predWbf, predb, bng, bnb, Ybf);
  dec_gemm_kernel<<<dim3(NTOK / 16 / 16, (TT * BB) / 32), 256, 0, stream>>>(
      Ybf, decWbf, decb, out);
  copy_hc_kernel<<<(2 * BB * NHID + 255) / 256, 256, 0, stream>>>(hState, cState, out);
}
